// GNN_53446573031651
// MI455X (gfx1250) — compile-verified, matrix operand order checked by reading, not run on hardware
//
#include <hip/hip_runtime.h>
#include <stdint.h>

// ---------------------------------------------------------------------------
// GCN: h = emb[x]; 2x (GEMM -> deg-normalized scatter-add -> +b -> relu);
// global_add_pool; pooled @ W3 + b3.
// GEMMs: v_wmma_f32_16x16x32_bf16, weights pre-transposed to [N,K] bf16 so
// both A and B fragments are contiguous 16B loads per lane.
// Scatter/pool: f32 global atomics into L2-resident buffers (51MB / 0.5MB
// vs 192MB L2 on MI455X).
// ---------------------------------------------------------------------------

typedef __attribute__((ext_vector_type(16))) __bf16 v16bf;
typedef __attribute__((ext_vector_type(8)))  __bf16 v8bf;
typedef __attribute__((ext_vector_type(8)))  float  v8f;

__device__ __forceinline__ unsigned short f2bf(float f) {
  unsigned int u = __float_as_uint(f);
  u += 0x7FFFu + ((u >> 16) & 1u);   // round-to-nearest-even
  return (unsigned short)(u >> 16);
}

__device__ __forceinline__ v16bf load_frag(const unsigned short* p) {
  // elements 0..7 = K kb+0..7 (contiguous), 8..15 = K kb+16..23 (contiguous)
  v8bf lo = *(const v8bf*)p;
  v8bf hi = *(const v8bf*)(p + 16);
  return __builtin_shufflevector(lo, hi, 0, 1, 2, 3, 4, 5, 6, 7,
                                         8, 9, 10, 11, 12, 13, 14, 15);
}

// ---- f32 -> bf16 bulk convert --------------------------------------------
__global__ void convert_bf16(const float* __restrict__ src,
                             unsigned short* __restrict__ dst, long long n) {
  long long i = (long long)blockIdx.x * blockDim.x + threadIdx.x;
  if (i < n) dst[i] = f2bf(src[i]);
}

// ---- W[K,N] f32 -> Wt[N,K] bf16, LDS-tiled transpose ---------------------
__global__ __launch_bounds__(256)
void transpose_convert_bf16(const float* __restrict__ W,
                            unsigned short* __restrict__ Wt, int K, int N) {
  __shared__ unsigned short tile[32][33];
  const int k0 = blockIdx.y * 32, n0 = blockIdx.x * 32;
  const int tx = threadIdx.x, ty = threadIdx.y;   // 32 x 8
#pragma unroll
  for (int r = ty; r < 32; r += 8) {
    int k = k0 + r, n = n0 + tx;
    tile[r][tx] = (k < K && n < N) ? f2bf(W[(size_t)k * N + n]) : (unsigned short)0;
  }
  __syncthreads();
#pragma unroll
  for (int r = ty; r < 32; r += 8) {
    int n = n0 + r, k = k0 + tx;
    if (n < N && k < K) Wt[(size_t)n * K + k] = tile[tx][r];
  }
}

// ---- embedding gather + convert ------------------------------------------
__global__ void embed_gather(const int* __restrict__ x,
                             const float* __restrict__ emb,
                             unsigned short* __restrict__ hbf,
                             int N, int D) {
  long long i = (long long)blockIdx.x * blockDim.x + threadIdx.x;
  if (i >= (long long)N * D) return;
  int n = (int)(i / D), d = (int)(i % D);
  hbf[i] = f2bf(emb[(size_t)x[n] * D + d]);
}

// ---- degree (with self loop) and rsqrt -----------------------------------
__global__ void deg_init(float* __restrict__ deg, int N) {
  int i = blockIdx.x * blockDim.x + threadIdx.x;
  if (i < N) deg[i] = 1.0f;                 // self loop
}
__global__ void deg_count(const int* __restrict__ col, float* __restrict__ deg, int E) {
  int i = blockIdx.x * blockDim.x + threadIdx.x;
  if (i < E) atomicAdd(&deg[col[i]], 1.0f); // exact: counts < 2^24
}
__global__ void deg_rsqrt(float* __restrict__ deg, int N) {
  int i = blockIdx.x * blockDim.x + threadIdx.x;
  if (i < N) deg[i] = rsqrtf(deg[i]);
}

// ---- WMMA bf16 GEMM: C[M,N] = A[M,K] @ Bt[N,K]^T (+bias) -----------------
// One wave per 16 x (NT*16) C tile; K fully unrolled; no in-kernel guards
// (tail tiles launched as a separate NT=1 grid) -> EXEC all-ones throughout.
template <int K, int NT>
__global__ __launch_bounds__(32)
void wmma_gemm_bf16(const unsigned short* __restrict__ A,   // [M,K] bf16
                    const unsigned short* __restrict__ Bt,  // [N,K] bf16
                    const float* __restrict__ bias,
                    float* __restrict__ C,                  // [M,N] f32
                    int N, int n_base) {
  const int lane = threadIdx.x & 31;
  const int m0   = blockIdx.y << 4;
  const int n0   = n_base + blockIdx.x * (NT * 16);
  const int half = lane >> 4;      // 0: K 0..7/16..23 ; 1: K 8..15/24..31
  const int lid  = lane & 15;      // A: row ; B/C: column
  const int kb   = half << 3;
  const unsigned short* Ap = A + (size_t)(m0 + lid) * K + kb;

  v8f acc[NT] = {};
#pragma unroll
  for (int k0 = 0; k0 < K; k0 += 32) {
    v16bf a = load_frag(Ap + k0);
#pragma unroll
    for (int t = 0; t < NT; ++t) {
      const unsigned short* Bp = Bt + (size_t)(n0 + (t << 4) + lid) * K + kb + k0;
      v16bf b = load_frag(Bp);
      acc[t] = __builtin_amdgcn_wmma_f32_16x16x32_bf16(
          false, a, false, b, (short)0, acc[t], false, false);
    }
  }
  // C/D layout: element r -> M = m0 + 8*half + r, N = n0 + 16t + lid
#pragma unroll
  for (int t = 0; t < NT; ++t) {
    const int   cc = n0 + (t << 4) + lid;
    const float bv = bias ? bias[cc] : 0.0f;
    float* Cp = C + (size_t)(m0 + (half << 3)) * N + cc;
#pragma unroll
    for (int r = 0; r < 8; ++r) Cp[(size_t)r * N] = acc[t][r] + bv;
  }
}

// ---- linS = lin * dinv[n]; acc = linS (self-loop contribution) -----------
__global__ void prescale_copy(float* __restrict__ lin, float* __restrict__ acc,
                              const float* __restrict__ dinv, int N, int F) {
  long long i = (long long)blockIdx.x * blockDim.x + threadIdx.x;
  long long tot = (long long)N * (F >> 2);
  if (i >= tot) return;
  int n  = (int)(i / (F >> 2));
  int f4 = (int)(i % (F >> 2)) << 2;
  float s = dinv[n];
  float4 v = *(const float4*)(lin + (size_t)n * F + f4);
  v.x *= s; v.y *= s; v.z *= s; v.w *= s;
  *(float4*)(lin + (size_t)n * F + f4) = v;
  *(float4*)(acc + (size_t)n * F + f4) = v;
}

// ---- edge scatter: acc[col] += linS[row] ---------------------------------
__global__ void edge_scatter(const int* __restrict__ row, const int* __restrict__ col,
                             const float* __restrict__ linS, float* __restrict__ acc,
                             int E, int F) {
  long long gid = (long long)blockIdx.x * blockDim.x + threadIdx.x;
  const int q = F >> 2;
  if (gid >= (long long)E * q) return;
  int e  = (int)(gid / q);
  int f4 = (int)(gid % q) << 2;
  int r = row[e], c = col[e];
  float4 v = *(const float4*)(linS + (size_t)r * F + f4);
  float* dst = acc + (size_t)c * F + f4;   // L2-resident accumulator
  atomicAdd(dst + 0, v.x);
  atomicAdd(dst + 1, v.y);
  atomicAdd(dst + 2, v.z);
  atomicAdd(dst + 3, v.w);
}

// ---- h = bf16(relu(dinv[n]*acc + b[f])) ----------------------------------
__global__ void finalize_relu_bf16(const float* __restrict__ acc,
                                   const float* __restrict__ dinv,
                                   const float* __restrict__ b,
                                   unsigned short* __restrict__ out, int N, int F) {
  long long i = (long long)blockIdx.x * blockDim.x + threadIdx.x;
  if (i >= (long long)N * F) return;
  int n = (int)(i / F), f = (int)(i % F);
  float v = dinv[n] * acc[i] + b[f];
  out[i] = f2bf(v > 0.0f ? v : 0.0f);
}

__global__ void zero_f32(float* __restrict__ p, long long n) {
  long long i = (long long)blockIdx.x * blockDim.x + threadIdx.x;
  if (i < n) p[i] = 0.0f;
}

// ---- layer-2 finalize fused with global_add_pool -------------------------
__global__ void finalize_pool(const float* __restrict__ acc,
                              const float* __restrict__ dinv,
                              const float* __restrict__ b,
                              const int* __restrict__ batch,
                              float* __restrict__ pooled, int N, int F) {
  long long i = (long long)blockIdx.x * blockDim.x + threadIdx.x;
  if (i >= (long long)N * F) return;
  int n = (int)(i / F), f = (int)(i % F);
  float v = dinv[n] * acc[i] + b[f];
  v = v > 0.0f ? v : 0.0f;
  atomicAdd(&pooled[(size_t)batch[n] * F + f], v);
}

// ---------------------------------------------------------------------------
extern "C" void kernel_launch(void* const* d_in, const int* in_sizes, int n_in,
                              void* d_out, int out_size, void* d_ws, size_t ws_size,
                              hipStream_t stream) {
  const int*   x     = (const int*)d_in[0];
  const int*   eidx  = (const int*)d_in[1];
  const int*   batch = (const int*)d_in[2];
  const float* emb   = (const float*)d_in[3];
  const float* W1    = (const float*)d_in[4];
  const float* b1    = (const float*)d_in[5];
  const float* W2    = (const float*)d_in[6];
  const float* b2    = (const float*)d_in[7];
  const float* W3    = (const float*)d_in[8];
  const float* b3    = (const float*)d_in[9];

  const int N = in_sizes[0];          // 50000 nodes
  const int E = in_sizes[1] / 2;      // 800000 edges
  const int H = in_sizes[5];          // 256
  const int D = in_sizes[4] / H;      // 128
  const int O = in_sizes[9];          // 50000
  const int G = out_size / O;         // 512 graphs
  const int* row = eidx;
  const int* col = eidx + E;

  // workspace carve (~168 MB total)
  char* wsb = (char*)d_ws; size_t off = 0;
  auto carve = [&](size_t bytes) -> void* {
    void* p = wsb + off;
    off = (off + bytes + 255) & ~(size_t)255;
    return p;
  };
  unsigned short* hbf   = (unsigned short*)carve((size_t)N * D * 2);
  unsigned short* w1t   = (unsigned short*)carve((size_t)D * H * 2);  // [H,D]
  unsigned short* w2t   = (unsigned short*)carve((size_t)H * H * 2);  // [H,H]
  unsigned short* w3t   = (unsigned short*)carve((size_t)H * O * 2);  // [O,H]
  float*          dinv  = (float*)carve((size_t)N * 4);
  float*          lin   = (float*)carve((size_t)N * H * 4);
  float*          acc   = (float*)carve((size_t)N * H * 4);
  unsigned short* h1bf  = (unsigned short*)carve((size_t)N * H * 2);
  float*          pool  = (float*)carve((size_t)G * H * 4);
  unsigned short* pbf   = (unsigned short*)carve((size_t)G * H * 2);
  (void)ws_size; (void)n_in;

  const int TB = 256;
  auto nb = [](long long n, int tb) { return (unsigned)((n + tb - 1) / tb); };
  dim3 tb28(32, 8);

  // weights -> bf16 transposed [Nout, K]; embeddings -> bf16
  transpose_convert_bf16<<<dim3((H + 31) / 32, (D + 31) / 32), tb28, 0, stream>>>(W1, w1t, D, H);
  transpose_convert_bf16<<<dim3((H + 31) / 32, (H + 31) / 32), tb28, 0, stream>>>(W2, w2t, H, H);
  transpose_convert_bf16<<<dim3((O + 31) / 32, (H + 31) / 32), tb28, 0, stream>>>(W3, w3t, H, O);
  embed_gather<<<nb((long long)N * D, TB), TB, 0, stream>>>(x, emb, hbf, N, D);

  // symmetric normalization terms (shared by both conv layers)
  deg_init <<<nb(N, TB), TB, 0, stream>>>(dinv, N);
  deg_count<<<nb(E, TB), TB, 0, stream>>>(col, dinv, E);
  deg_rsqrt<<<nb(N, TB), TB, 0, stream>>>(dinv, N);

  dim3 g1(H / 64, N / 16);   // H==256: no tail

  // ---- GCN layer 1 (K = D = 128) ----
  wmma_gemm_bf16<128, 4><<<g1, 32, 0, stream>>>(hbf, w1t, nullptr, lin, H, 0);
  prescale_copy<<<nb((long long)N * (H / 4), TB), TB, 0, stream>>>(lin, acc, dinv, N, H);
  edge_scatter<<<nb((long long)E * (H / 4), TB), TB, 0, stream>>>(row, col, lin, acc, E, H);
  finalize_relu_bf16<<<nb((long long)N * H, TB), TB, 0, stream>>>(acc, dinv, b1, h1bf, N, H);

  // ---- GCN layer 2 (K = H = 256, finalize fused with pooling) ----
  wmma_gemm_bf16<256, 4><<<g1, 32, 0, stream>>>(h1bf, w2t, nullptr, lin, H, 0);
  prescale_copy<<<nb((long long)N * (H / 4), TB), TB, 0, stream>>>(lin, acc, dinv, N, H);
  edge_scatter<<<nb((long long)E * (H / 4), TB), TB, 0, stream>>>(row, col, lin, acc, E, H);
  zero_f32<<<nb((long long)G * H, TB), TB, 0, stream>>>(pool, (long long)G * H);
  finalize_pool<<<nb((long long)N * H, TB), TB, 0, stream>>>(acc, dinv, b2, batch, pool, N, H);

  // ---- classifier GEMM: [G,H] x [H,O] + b3 ----
  convert_bf16<<<nb((long long)G * H, TB), TB, 0, stream>>>(pool, pbf, (long long)G * H);
  const int fullO = O / 64;                 // 781 full 64-wide groups
  const int remO  = (O - fullO * 64) / 16;  // 1 trailing 16-wide subtile
  if (fullO > 0)
    wmma_gemm_bf16<256, 4><<<dim3(fullO, G / 16), 32, 0, stream>>>(
        pbf, w3t, b3, (float*)d_out, O, 0);
  if (remO > 0)
    wmma_gemm_bf16<256, 1><<<dim3(remO, G / 16), 32, 0, stream>>>(
        pbf, w3t, b3, (float*)d_out, O, fullO * 64);
}
